// RNN_test_55860344651963
// MI455X (gfx1250) — compile-verified, matrix operand order checked by reading, not run on hardware
//
#include <hip/hip_runtime.h>
#include <hip/hip_fp16.h>
#include <hip/hip_bf16.h>

// ---------------------------------------------------------------------------
// DH-SNN forward, persistent kernel, CDNA5 (gfx1250) wave32 + WMMA f16.
// Grid = 16 workgroups (one per 16-row batch tile), 256 threads (8 waves).
// All recurrent state on-chip for all 1000 steps; WMMA 16x16x32 f16->f32.
// ---------------------------------------------------------------------------

typedef __attribute__((ext_vector_type(16))) _Float16 v16h;
typedef __attribute__((ext_vector_type(8)))  _Float16 v8h;
typedef __attribute__((ext_vector_type(8)))  float    v8f;

#define IN_DIM  76
#define N1      50
#define N2      100
#define OUT_DIM 2
#define SEQ     1000
#define BT      16          // batch tile rows per workgroup

// padded GEMM geometry
#define K1P  128            // 126 -> 128  (4 K-chunks of 32)
#define NC1  200            // N1*BRANCH
#define NT1  13             // ceil(208/16)
#define K2P  160            // 150 -> 160  (5 K-chunks)
#define NC2  400
#define NT2  25
#define KLP  224            // 200 -> 224  (7 K-chunks)
#define NLP  112            // 100 -> 112  (7 tiles)

// f16 weight arena layout in d_ws (element counts)
#define W1H_OFF 0
#define W1H_CNT (208 * K1P)             // 26624
#define W2H_OFF (W1H_OFF + W1H_CNT)
#define W2H_CNT (NC2 * K2P)             // 64000
#define WXH_OFF (W2H_OFF + W2H_CNT)
#define WLH_CNT (NLP * KLP)             // 25088
#define WMH_OFF (WXH_OFF + WLH_CNT)
#define WAH_OFF (WMH_OFF + WLH_CNT)

__device__ __forceinline__ float sigmoidf(float x) {
    return 1.0f / (1.0f + __expf(-x));
}

// Load one 16x32 (or mirrored 32x16-as-NxK) f16 WMMA fragment.
// lane l: row = caller-provided, klo = 32*kc + (l&16 ? 8 : 0);
// vector elems 0..7 = K[klo..klo+7], elems 8..15 = K[klo+16..klo+23].
__device__ __forceinline__ v16h load_frag(const _Float16* base, int row,
                                          int kpad, int kc, int lane) {
    int klo = kc * 32 + ((lane & 16) ? 8 : 0);
    const v8h* p = reinterpret_cast<const v8h*>(base + row * kpad + klo);
    v8h lo = p[0];
    v8h hi = p[2];  // +16 halfs
    return __builtin_shufflevector(lo, hi, 0, 1, 2, 3, 4, 5, 6, 7,
                                   8, 9, 10, 11, 12, 13, 14, 15);
}

__device__ __forceinline__ v8f wmma32(v8f c, v16h a, v16h b) {
    // v_wmma_f32_16x16x32_f16  D = A*B + C
    return __builtin_amdgcn_wmma_f32_16x16x32_f16(false, a, false, b,
                                                  (short)0, c, false, false);
}

// ---------------------------------------------------------------------------
// Prep: fp32 -> f16 weight conversion, dendritic branch mask applied,
// padded to WMMA tile geometry, stored as W[n][k] row-major (== B^T).
// ---------------------------------------------------------------------------
__global__ __launch_bounds__(256) void snn_prep(const float* __restrict__ W1,
                                                const float* __restrict__ W2,
                                                const float* __restrict__ Wx,
                                                const float* __restrict__ Wm,
                                                const float* __restrict__ Wa,
                                                _Float16* __restrict__ ws) {
    int i = blockIdx.x * blockDim.x + threadIdx.x;
    if (i < W1H_CNT) {                               // W1 masked: seg=32
        int n = i / K1P, k = i - n * K1P;
        float v = 0.0f;
        if (n < NC1 && k < 126) {
            int br = n & 3, lo = br * 32, hi = lo + 32 < 126 ? lo + 32 : 126;
            if (k >= lo && k < hi) v = W1[n * 126 + k];
        }
        ws[W1H_OFF + i] = (_Float16)v;
        return;
    }
    i -= W1H_CNT;
    if (i < W2H_CNT) {                               // W2 masked: seg=38
        int n = i / K2P, k = i - n * K2P;
        float v = 0.0f;
        if (n < NC2 && k < 150) {
            int br = n & 3, lo = br * 38, hi = lo + 38 < 150 ? lo + 38 : 150;
            if (k >= lo && k < hi) v = W2[n * 150 + k];
        }
        ws[W2H_OFF + i] = (_Float16)v;
        return;
    }
    i -= W2H_CNT;
    int r = i / WLH_CNT;                             // 0:Wx 1:Wm 2:Wa
    if (r < 3) {
        int j = i - r * WLH_CNT;
        int n = j / KLP, k = j - n * KLP;
        const float* src = (r == 0) ? Wx : ((r == 1) ? Wm : Wa);
        float v = (n < N2 && k < 200) ? src[n * 200 + k] : 0.0f;
        ws[WXH_OFF + r * WLH_CNT + j] = (_Float16)v;
    }
}

// ---------------------------------------------------------------------------
// Persistent forward kernel.
// ---------------------------------------------------------------------------
struct SM {
    _Float16 W1l[W1H_CNT];      // 53 KB, async-loaded once
    _Float16 A1[BT * K1P];      // [x_t | s1]
    _Float16 A2[BT * K2P];      // [s1 | s2]
    _Float16 AL[BT * KLP];      // liquid concat buffer
    float    proj[BT * 416];    // D-fragment landing zone (max 400 cols)
    float    dxf[BT * NLP];
    float    tmb[BT * NLP];
    float    m1[BT * N1], s1[BT * N1];
    float    m2[BT * N2];
    float    lm[BT * N2], ls[BT * N2], lb[BT * N2];
    float    md[BT * OUT_DIM], acc[BT * OUT_DIM];
    float    alph1[N1], alph2[N2];
    float    Wdl[OUT_DIM * N2], bdl[OUT_DIM], alphd[OUT_DIM];
};

__global__ __launch_bounds__(256) void snn_forward(
    const float* __restrict__ x,
    const float* __restrict__ b1,  const float* __restrict__ tau_m1,
    const float* __restrict__ tau_n1,
    const float* __restrict__ b2,  const float* __restrict__ tau_m2,
    const float* __restrict__ tau_n2,
    const float* __restrict__ bx,  const float* __restrict__ bm,
    const float* __restrict__ ba,
    const float* __restrict__ Wd,  const float* __restrict__ bd,
    const float* __restrict__ tau_md,
    const _Float16* __restrict__ W1h, const _Float16* __restrict__ W2h,
    const _Float16* __restrict__ Wxh, const _Float16* __restrict__ Wmh,
    const _Float16* __restrict__ Wah,
    float* __restrict__ out) {
    __shared__ SM sm;

    const int tid  = threadIdx.x;
    const int lane = tid & 31;
    const int wave = tid >> 5;
    const int col  = lane & 15;
    const int mrow = (lane & 16) ? 8 : 0;
    const int m0   = blockIdx.x * BT;                // batch tile origin

    // ---- one-time: async DMA W1 (f16) into LDS -------------------------
    for (int i = tid; i < W1H_CNT / 8; i += 256) {   // 16B per op
        unsigned lds = (unsigned)(unsigned long long)(&sm.W1l[i * 8]);
        unsigned long long g = (unsigned long long)(const void*)(W1h + i * 8);
        asm volatile("global_load_async_to_lds_b128 %0, %1, off"
                     :: "v"(lds), "v"(g) : "memory");
    }
    asm volatile("s_wait_asynccnt 0" ::: "memory");

    // ---- one-time: zero state / load small params ----------------------
    for (int i = tid; i < BT * N1; i += 256) { sm.m1[i] = 0.f; sm.s1[i] = 0.f; }
    for (int i = tid; i < BT * N2; i += 256) {
        sm.m2[i] = 0.f; sm.lm[i] = 0.f; sm.ls[i] = 0.f; sm.lb[i] = 1.6f;
    }
    for (int i = tid; i < BT * K1P; i += 256) sm.A1[i] = (_Float16)0.f;
    for (int i = tid; i < BT * K2P; i += 256) sm.A2[i] = (_Float16)0.f;
    for (int i = tid; i < BT * KLP; i += 256) sm.AL[i] = (_Float16)0.f;
    for (int i = tid; i < N1; i += 256) sm.alph1[i] = sigmoidf(tau_m1[i]);
    for (int i = tid; i < N2; i += 256) sm.alph2[i] = sigmoidf(tau_m2[i]);
    for (int i = tid; i < OUT_DIM * N2; i += 256) sm.Wdl[i] = Wd[i];
    if (tid < OUT_DIM) { sm.bdl[tid] = bd[tid]; sm.alphd[tid] = sigmoidf(tau_md[tid]); }
    for (int i = tid; i < BT * OUT_DIM; i += 256) { sm.md[i] = 0.f; sm.acc[i] = 0.f; }

    // ---- per-wave persistent registers (β, bias, dendrite accumulators)
    const v8f vzero = {0.f, 0.f, 0.f, 0.f, 0.f, 0.f, 0.f, 0.f};
    float bet1[2], bia1[2]; v8f d1[2];
    #pragma unroll
    for (int s = 0; s < 2; ++s) {
        bet1[s] = 0.f; bia1[s] = 0.f; d1[s] = vzero;
        int tn = wave + 8 * s;
        if (tn < NT1) {
            int c = tn * 16 + col;
            if (c < NC1) { bet1[s] = sigmoidf(tau_n1[c]); bia1[s] = b1[c]; }
        }
    }
    float bet2[4], bia2[4]; v8f d2[4];
    #pragma unroll
    for (int s = 0; s < 4; ++s) {
        bet2[s] = 0.f; bia2[s] = 0.f; d2[s] = vzero;
        int tn = wave + 8 * s;
        if (tn < NT2) {
            int c = tn * 16 + col;
            if (c < NC2) { bet2[s] = sigmoidf(tau_n2[c]); bia2[s] = b2[c]; }
        }
    }
    float biaX = 0.f, biaM = 0.f, biaA = 0.f;
    if (wave < 7) {
        int c = wave * 16 + col;
        if (c < N2) { biaX = bx[c]; biaM = bm[c]; biaA = ba[c]; }
    }
    __syncthreads();

    // =================== time loop (fully on-chip) ======================
    for (int t = 0; t < SEQ; ++t) {
        // -- load x_t into A1[:, 0:76] (f32 -> f16)
        for (int i = tid; i < BT * IN_DIM; i += 256) {
            int b = i / IN_DIM, k = i - b * IN_DIM;
            sm.A1[b * K1P + k] =
                (_Float16)x[(size_t)(m0 + b) * (SEQ * IN_DIM) + t * IN_DIM + k];
        }
        __syncthreads();

        // -- L1 GEMM: A1[16,128] x W1^T (LDS), dendrite update in regs
        #pragma unroll
        for (int s = 0; s < 2; ++s) {
            int tn = wave + 8 * s;
            if (tn < NT1) {
                v8f c = vzero;
                #pragma unroll
                for (int kc = 0; kc < 4; ++kc) {
                    v16h a  = load_frag(sm.A1, col, K1P, kc, lane);
                    v16h bf = load_frag(sm.W1l, tn * 16 + col, K1P, kc, lane);
                    c = wmma32(c, a, bf);
                }
                float bt = bet1[s];
                #pragma unroll
                for (int v = 0; v < 8; ++v) {
                    float dn = bt * d1[s][v] + (1.f - bt) * (c[v] + bia1[s]);
                    d1[s][v] = dn;
                    sm.proj[(v + mrow) * 416 + tn * 16 + col] = dn;
                }
            }
        }
        __syncthreads();

        // -- L1 neuron update (branch sum, LIF, spike)
        for (int i = tid; i < BT * N1; i += 256) {
            int b = i / N1, n = i - b * N1;
            const float* pr = &sm.proj[b * 416 + 4 * n];
            float sum = pr[0] + pr[1] + pr[2] + pr[3];
            float al  = sm.alph1[n];
            float so  = sm.s1[i];
            float mem = sm.m1[i] * al + (1.f - al) * sum - 1.0f * so;
            float sp  = (mem - 1.0f) > 0.f ? 1.f : 0.f;
            sm.m1[i] = mem; sm.s1[i] = sp;
            _Float16 h = (_Float16)sp;
            sm.A1[b * K1P + IN_DIM + n] = h;   // next step's L1 input
            sm.A2[b * K2P + n]          = h;   // this step's L2 input
        }
        __syncthreads();

        // -- L2 GEMM: A2[16,160] x W2^T (global/L2-resident)
        #pragma unroll
        for (int s = 0; s < 4; ++s) {
            int tn = wave + 8 * s;
            if (tn < NT2) {
                __builtin_prefetch(W2h + (tn * 16 + col) * K2P, 0, 1);
                v8f c = vzero;
                #pragma unroll
                for (int kc = 0; kc < 5; ++kc) {
                    v16h a  = load_frag(sm.A2, col, K2P, kc, lane);
                    v16h bf = load_frag(W2h, tn * 16 + col, K2P, kc, lane);
                    c = wmma32(c, a, bf);
                }
                float bt = bet2[s];
                #pragma unroll
                for (int v = 0; v < 8; ++v) {
                    float dn = bt * d2[s][v] + (1.f - bt) * (c[v] + bia2[s]);
                    d2[s][v] = dn;
                    sm.proj[(v + mrow) * 416 + tn * 16 + col] = dn;
                }
            }
        }
        __syncthreads();

        // -- L2 neuron update
        for (int i = tid; i < BT * N2; i += 256) {
            int b = i / N2, n = i - b * N2;
            const float* pr = &sm.proj[b * 416 + 4 * n];
            float sum = pr[0] + pr[1] + pr[2] + pr[3];
            float al  = sm.alph2[n];
            float so  = (float)sm.A2[b * K2P + N1 + n];   // previous s2
            float mem = sm.m2[i] * al + (1.f - al) * sum - 1.0f * so;
            float sp  = (mem - 1.0f) > 0.f ? 1.f : 0.f;
            sm.m2[i] = mem;
            _Float16 h = (_Float16)sp;
            sm.A2[b * K2P + N1 + n] = h;       // next step's L2 input
            sm.AL[b * KLP + n]      = h;       // liquid x_t
        }
        __syncthreads();

        // -- liquid: AL = [s2 | ls_prev]
        for (int i = tid; i < BT * N2; i += 256) {
            int b = i / N2, n = i - b * N2;
            sm.AL[b * KLP + N2 + n] = (_Float16)sm.ls[i];
        }
        __syncthreads();

        // -- dx = AL @ Wx^T + bx
        if (wave < 7) {
            v8f c = vzero;
            #pragma unroll
            for (int kc = 0; kc < 7; ++kc) {
                v16h a  = load_frag(sm.AL, col, KLP, kc, lane);
                v16h bf = load_frag(Wxh, wave * 16 + col, KLP, kc, lane);
                c = wmma32(c, a, bf);
            }
            #pragma unroll
            for (int v = 0; v < 8; ++v)
                sm.dxf[(v + mrow) * NLP + wave * 16 + col] = c[v] + biaX;
        }
        __syncthreads();

        // -- AL = [dx | lm]
        for (int i = tid; i < BT * N2; i += 256) {
            int b = i / N2, n = i - b * N2;
            sm.AL[b * KLP + n]      = (_Float16)sm.dxf[b * NLP + n];
            sm.AL[b * KLP + N2 + n] = (_Float16)sm.lm[i];
        }
        __syncthreads();

        // -- tauM = sigmoid(AL @ Wm^T + bm)
        if (wave < 7) {
            v8f c = vzero;
            #pragma unroll
            for (int kc = 0; kc < 7; ++kc) {
                v16h a  = load_frag(sm.AL, col, KLP, kc, lane);
                v16h bf = load_frag(Wmh, wave * 16 + col, KLP, kc, lane);
                c = wmma32(c, a, bf);
            }
            #pragma unroll
            for (int v = 0; v < 8; ++v)
                sm.tmb[(v + mrow) * NLP + wave * 16 + col] = sigmoidf(c[v] + biaM);
        }
        __syncthreads();

        // -- AL = [dx | lb]
        for (int i = tid; i < BT * N2; i += 256) {
            int b = i / N2, n = i - b * N2;
            sm.AL[b * KLP + N2 + n] = (_Float16)sm.lb[i];
        }
        __syncthreads();

        // -- tauA = sigmoid(AL @ Wa^T + ba) -> proj
        if (wave < 7) {
            v8f c = vzero;
            #pragma unroll
            for (int kc = 0; kc < 7; ++kc) {
                v16h a  = load_frag(sm.AL, col, KLP, kc, lane);
                v16h bf = load_frag(Wah, wave * 16 + col, KLP, kc, lane);
                c = wmma32(c, a, bf);
            }
            #pragma unroll
            for (int v = 0; v < 8; ++v)
                sm.proj[(v + mrow) * 416 + wave * 16 + col] = sigmoidf(c[v] + biaA);
        }
        __syncthreads();

        // -- liquid cell update (adaptive threshold LIF)
        for (int i = tid; i < BT * N2; i += 256) {
            int b = i / N2, n = i - b * N2;
            float tauA = sm.proj[b * 416 + n];
            float so   = sm.ls[i];
            float bn   = tauA * sm.lb[i] + (1.f - tauA) * so;
            float Bv   = 1.6f + 1.8f * bn;
            float tauM = sm.tmb[b * NLP + n];
            float mem  = sm.lm[i] * tauM + (1.f - tauM) * sm.dxf[b * NLP + n]
                         - Bv * so;
            float sp   = (mem - Bv) > 0.f ? 1.f : 0.f;
            sm.lm[i] = mem; sm.ls[i] = sp; sm.lb[i] = bn;
        }
        __syncthreads();

        // -- readout integrator + accumulation (t >= 1)
        if (tid < BT * OUT_DIM) {
            int b = tid >> 1, o = tid & 1;
            float s = sm.bdl[o];
            for (int j = 0; j < N2; ++j)
                s += sm.ls[b * N2 + j] * sm.Wdl[o * N2 + j];
            float a = sm.alphd[o];
            float m = sm.md[b * OUT_DIM + o] * a + (1.f - a) * s;
            sm.md[b * OUT_DIM + o] = m;
            if (t > 0) sm.acc[b * OUT_DIM + o] += m;
        }
        __syncthreads();
    }

    if (tid < BT * OUT_DIM) {
        int b = tid >> 1, o = tid & 1;
        out[(size_t)(m0 + b) * OUT_DIM + o] = sm.acc[b * OUT_DIM + o];
    }
}

// ---------------------------------------------------------------------------
extern "C" void kernel_launch(void* const* d_in, const int* in_sizes, int n_in,
                              void* d_out, int out_size, void* d_ws, size_t ws_size,
                              hipStream_t stream) {
    const float* x      = (const float*)d_in[0];
    const float* W1     = (const float*)d_in[1];
    const float* b1     = (const float*)d_in[2];
    const float* tau_m1 = (const float*)d_in[3];
    const float* tau_n1 = (const float*)d_in[4];
    const float* W2     = (const float*)d_in[5];
    const float* b2     = (const float*)d_in[6];
    const float* tau_m2 = (const float*)d_in[7];
    const float* tau_n2 = (const float*)d_in[8];
    const float* Wx     = (const float*)d_in[9];
    const float* bx     = (const float*)d_in[10];
    const float* Wm     = (const float*)d_in[11];
    const float* bm     = (const float*)d_in[12];
    const float* Wa     = (const float*)d_in[13];
    const float* ba     = (const float*)d_in[14];
    const float* Wd     = (const float*)d_in[15];
    const float* bd     = (const float*)d_in[16];
    const float* tau_md = (const float*)d_in[17];

    _Float16* ws = (_Float16*)d_ws;
    const int total = W1H_CNT + W2H_CNT + 3 * WLH_CNT;
    snn_prep<<<(total + 255) / 256, 256, 0, stream>>>(W1, W2, Wx, Wm, Wa, ws);

    snn_forward<<<256 / BT, 256, 0, stream>>>(
        x, b1, tau_m1, tau_n1, b2, tau_m2, tau_n2, bx, bm, ba, Wd, bd, tau_md,
        ws + W1H_OFF, ws + W2H_OFF, ws + WXH_OFF, ws + WMH_OFF, ws + WAH_OFF,
        (float*)d_out);
}
